// ImSituationHandler_41798621725059
// MI455X (gfx1250) — matrix operand order repaired and checked by reading, not compiled, not run on hardware
//
#include <hip/hip_runtime.h>

// ---------------- types ----------------
typedef _Float16 half_t;
typedef __attribute__((ext_vector_type(16))) _Float16 v16h;
typedef __attribute__((ext_vector_type(8)))  _Float16 v8h;
typedef __attribute__((ext_vector_type(8)))  float    v8f;
typedef int v4i __attribute__((vector_size(16)));
typedef __attribute__((address_space(1))) v4i* gptr4;   // global int4*
typedef __attribute__((address_space(3))) v4i* lptr4;   // LDS int4*

// CDNA5 async global->LDS path (guarded; falls back to plain LDS stores)
#if defined(__has_builtin)
#  if __has_builtin(__builtin_amdgcn_global_load_async_to_lds_b128)
#    define HAVE_ASYNC_LDS 1
#  endif
#endif

__device__ __forceinline__ void async_wait0()
{
#if defined(HAVE_ASYNC_LDS)
#  if __has_builtin(__builtin_amdgcn_s_wait_asynccnt)
    __builtin_amdgcn_s_wait_asynccnt(0);
#  else
    asm volatile("s_wait_asynccnt 0x0" ::: "memory");
#  endif
#endif
}

// ---------------- problem dims ----------------
#define BB   256
#define RR   6
#define LL   10
#define KK   49
#define EE   300
#define EP   320          // E padded to multiple of 32
#define HH   512
#define VV   2001
#define VP   2048         // V padded to multiple of 64
#define NN   (BB*RR)      // 1536
#define NK   (NN*KK)      // 75264
#define NL   (NN*LL)      // 15360
#define G4   (4*HH)       // 2048
#define PAIR (RR*RR)      // 36
#define NP   (BB*PAIR)    // 9216

__device__ __forceinline__ float sigf(float x) { return 1.0f / (1.0f + __expf(-x)); }

// =====================================================================
// Tiled WMMA GEMM:  C[M,N] = act( A[M,K] * B[N,K]^T + bias )
// A,B f16 row-major, ld = Kd (Kd % 32 == 0).  M % 128 == 0, Npad % 64 == 0.
// Block = 128 threads (4 waves) -> 128x64 tile; each wave 32x64 with
// 8 v_wmma_f32_16x16x32_f16 per K-step.  B panel (shared by all 4 waves)
// is staged into LDS in K-chunks via async global->LDS copies.
// =====================================================================
__device__ __forceinline__ void gemm_store(v8f acc, int col, int rbase, int Nreal,
                                           int ldc, const float* bias, int act,
                                           float* Cf, half_t* Ch)
{
    if (col >= Nreal) return;
    float bv = bias ? bias[col] : 0.0f;
#pragma unroll
    for (int v = 0; v < 8; ++v) {
        float x = acc[v] + bv;
        if (act) x = fmaxf(x, 0.0f);
        size_t o = (size_t)(rbase + v) * (size_t)ldc + (size_t)col;
        if (Cf) Cf[o] = x;
        if (Ch) Ch[o] = (half_t)x;
    }
}

__device__ __forceinline__ v16h frag16(const half_t* p0, const half_t* p1)
{
    v8h a0 = *(const v8h*)p0;
    v8h a1 = *(const v8h*)p1;
    return __builtin_shufflevector(a0, a1, 0,1,2,3,4,5,6,7,8,9,10,11,12,13,14,15);
}

__global__ __launch_bounds__(128)
void gemm_nt_wmma(const half_t* __restrict__ A, const half_t* __restrict__ Bm,
                  float* __restrict__ Cf, half_t* __restrict__ Ch,
                  const float* __restrict__ bias,
                  int M, int Nreal, int Kd, int ldc, int act)
{
    extern __shared__ half_t Bl[];           // 64 * (KCH+8) halves
    const int tid  = threadIdx.x;
    const int lane = tid & 31;
    const int wave = tid >> 5;
    const int col0 = blockIdx.x * 64;
    const int row0 = blockIdx.y * 128 + wave * 32;
    const int lr   = lane & 15;
    const int sel  = lane >> 4;              // half-wave select

    const int KCH = (Kd > 320) ? 256 : Kd;   // K chunk staged in LDS
    const int ldl = KCH + 8;                 // padded LDS row stride (bank-spread)

    const half_t* Arow0 = A + (size_t)(row0 + lr) * (size_t)Kd;
    const half_t* Arow1 = A + (size_t)(row0 + 16 + lr) * (size_t)Kd;

    v8f accA[4] = {{0}, {0}, {0}, {0}};
    v8f accB[4] = {{0}, {0}, {0}, {0}};

    for (int kb = 0; kb < Kd; kb += KCH) {
        if (kb) __syncthreads();             // previous chunk's reads done

        // ---- stage B chunk: 64 columns x KCH halves ----
        {
            const int cw = KCH / 8;          // 16B chunks per row
            for (int i = tid; i < 64 * cw; i += 128) {
                int r = i / cw;
                int o = (i - r * cw) * 8;
                const half_t* src = Bm + (size_t)(col0 + r) * (size_t)Kd + kb + o;
                half_t*       dst = Bl + (size_t)r * ldl + o;
#if defined(HAVE_ASYNC_LDS)
                __builtin_amdgcn_global_load_async_to_lds_b128((gptr4)src, (lptr4)dst, 0, 0);
#else
                *(v8h*)dst = *(const v8h*)src;
#endif
            }
        }
        async_wait0();
        __syncthreads();

        // ---- compute over the staged chunk ----
        for (int k0 = 0; k0 < KCH; k0 += 32) {
            // A 16x32 f16 fragments: lane<16 -> K {0..7,16..23}; lane>=16 -> {8..15,24..31}
            const int ka = kb + k0 + sel * 8;
            v16h af0 = frag16(Arow0 + ka, Arow0 + ka + 16);
            v16h af1 = frag16(Arow1 + ka, Arow1 + ka + 16);

#pragma unroll
            for (int j = 0; j < 4; ++j) {
                // B 32x16 fragment from LDS: lane<16 -> K {0..7,16..23}; lane>=16 -> {8..15,24..31}
                const half_t* bp = Bl + (size_t)(j * 16 + lr) * ldl + sel * 8 + k0;
                v16h bf = frag16(bp, bp + 16);
                accA[j] = __builtin_amdgcn_wmma_f32_16x16x32_f16(false, af0, false, bf, (short)0, accA[j], false, false);
                accB[j] = __builtin_amdgcn_wmma_f32_16x16x32_f16(false, af1, false, bf, (short)0, accB[j], false, false);
            }
        }
    }

    // C/D layout: VGPR v, lanes 0-15 -> M=v, lanes 16-31 -> M=v+8; N = lane&15
    const int rbA = row0 + sel * 8;
    const int rbB = row0 + 16 + sel * 8;
#pragma unroll
    for (int j = 0; j < 4; ++j) {
        gemm_store(accA[j], col0 + j * 16 + lr, rbA, Nreal, ldc, bias, act, Cf, Ch);
        gemm_store(accB[j], col0 + j * 16 + lr, rbB, Nreal, ldc, bias, act, Cf, Ch);
    }
}

// =====================================================================
// Pointwise / helper kernels
// =====================================================================
__global__ void cvt_pad_kernel(const float* __restrict__ src, half_t* __restrict__ dst,
                               int rows, int cols, int cols_pad, long long total)
{
    long long i = (long long)blockIdx.x * blockDim.x + threadIdx.x;
    if (i >= total) return;
    int c = (int)(i % cols_pad);
    long long r = i / cols_pad;
    float v = 0.0f;
    if (r < rows && c < cols) v = src[r * cols + c];
    dst[i] = (half_t)v;
}

__global__ void embed_kernel(const int* __restrict__ toks, const float* __restrict__ emb,
                             half_t* __restrict__ dst, long long total)
{
    long long i = (long long)blockIdx.x * blockDim.x + threadIdx.x;
    if (i >= total) return;
    int e = (int)(i % EP);
    long long row = i / EP;
    int t = toks[row];
    float v = (e < EE) ? emb[(size_t)t * EE + e] : 0.0f;
    dst[i] = (half_t)v;
}

__global__ void zero16_kernel(half_t* p, long long n)
{
    long long i = (long long)blockIdx.x * blockDim.x + threadIdx.x;
    if (i < n) p[i] = (half_t)0.0f;
}
__global__ void zero32_kernel(float* p, long long n)
{
    long long i = (long long)blockIdx.x * blockDim.x + threadIdx.x;
    if (i < n) p[i] = 0.0f;
}

// LSTM gate math (PyTorch order i,f,g,o). gx rows are (n*LL + tt).
__global__ __launch_bounds__(256)
void lstm_gate_kernel(const half_t* __restrict__ gx, const float* __restrict__ gh,
                      const float* __restrict__ bih, const float* __restrict__ bhh,
                      float* __restrict__ c, half_t* __restrict__ h, int tt)
{
    int idx = blockIdx.x * blockDim.x + threadIdx.x;     // 0 .. NN*HH-1
    if (idx >= NN * HH) return;
    int n  = idx >> 9;
    int hh = idx & (HH - 1);
    size_t bx = ((size_t)n * LL + tt) * G4;
    size_t bh = (size_t)n * G4;
    float gi = (float)gx[bx + hh]            + gh[bh + hh]            + bih[hh]            + bhh[hh];
    float gf = (float)gx[bx + HH + hh]       + gh[bh + HH + hh]       + bih[HH + hh]       + bhh[HH + hh];
    float gg = (float)gx[bx + 2 * HH + hh]   + gh[bh + 2 * HH + hh]   + bih[2 * HH + hh]   + bhh[2 * HH + hh];
    float go = (float)gx[bx + 3 * HH + hh]   + gh[bh + 3 * HH + hh]   + bih[3 * HH + hh]   + bhh[3 * HH + hh];
    float cn = sigf(gf) * c[idx] + sigf(gi) * tanhf(gg);
    c[idx] = cn;
    h[idx] = (half_t)(sigf(go) * tanhf(cn));
}

__global__ void hcat_kernel(const half_t* __restrict__ hf, const half_t* __restrict__ hb,
                            half_t* __restrict__ o)
{
    long long i = (long long)blockIdx.x * blockDim.x + threadIdx.x;
    if (i >= (long long)NN * 1024) return;
    int c = (int)(i & 1023);
    long long n = i >> 10;
    o[i] = (c < HH) ? hf[n * HH + c] : hb[n * HH + (c - HH)];
}

// att_logits[n,k] = sum_h vproj[n,k,h]*qproj[n,h]*oW[h] + ob  (one wave per row)
__global__ __launch_bounds__(256)
void attlogit_kernel(const half_t* __restrict__ vproj, const half_t* __restrict__ qproj,
                     const float* __restrict__ oW, const float* __restrict__ ob,
                     float* __restrict__ logits)
{
    int gw = blockIdx.x * 8 + (threadIdx.x >> 5);
    if (gw >= NK) return;
    int lane = threadIdx.x & 31;
    int n = gw / KK;
    const half_t* vp = vproj + (size_t)gw * HH;
    const half_t* qp = qproj + (size_t)n * HH;
    float s = 0.0f;
    for (int h = lane; h < HH; h += 32)
        s += (float)vp[h] * (float)qp[h] * oW[h];
#pragma unroll
    for (int off = 16; off; off >>= 1) s += __shfl_xor(s, off, 32);
    if (lane == 0) logits[gw] = s + ob[0];
}

// softmax over K regions + v_emb[n,h] = sum_k att[k]*img[n,k,h]
__global__ __launch_bounds__(256)
void softmax_vemb_kernel(const float* __restrict__ logits, const float* __restrict__ img,
                         half_t* __restrict__ vemb)
{
    int n = blockIdx.x;
    __shared__ float att[KK];
    if (threadIdx.x == 0) {
        float m = -1e30f;
        for (int k = 0; k < KK; ++k) m = fmaxf(m, logits[n * KK + k]);
        float s = 0.0f;
        for (int k = 0; k < KK; ++k) { float e = __expf(logits[n * KK + k] - m); att[k] = e; s += e; }
        float inv = 1.0f / s;
        for (int k = 0; k < KK; ++k) att[k] *= inv;
    }
    __syncthreads();
    for (int h = threadIdx.x; h < HH; h += 256) {
        float acc = 0.0f;
        const float* ip = img + (size_t)n * KK * HH + h;
#pragma unroll 7
        for (int k = 0; k < KK; ++k) acc += att[k] * ip[(size_t)k * HH];
        vemb[(size_t)n * HH + h] = (half_t)acc;
    }
}

__global__ void mul_kernel(const half_t* __restrict__ a, const half_t* __restrict__ b,
                           half_t* __restrict__ o, long long n)
{
    long long i = (long long)blockIdx.x * blockDim.x + threadIdx.x;
    if (i < n) o[i] = (half_t)((float)a[i] * (float)b[i]);
}

// cat[(b*6+i)*6+j] = [ joint[b*6+j] , joint[b*6+i] ]
__global__ void cat_kernel(const half_t* __restrict__ joint, half_t* __restrict__ cat)
{
    long long i = (long long)blockIdx.x * blockDim.x + threadIdx.x;
    if (i >= (long long)NP * 1024) return;
    int c = (int)(i & 1023);
    long long row = i >> 10;
    int b  = (int)(row / PAIR);
    int p  = (int)(row % PAIR);
    int ii = p / RR, jj = p % RR;
    half_t v = (c < HH) ? joint[((size_t)(b * RR + jj)) * HH + c]
                        : joint[((size_t)(b * RR + ii)) * HH + (c - HH)];
    cat[i] = v;
}

__global__ void gsum_kernel(const half_t* __restrict__ g, half_t* __restrict__ o)
{
    int i = blockIdx.x * blockDim.x + threadIdx.x;        // 0 .. BB*HH-1
    if (i >= BB * HH) return;
    int b = i >> 9, h = i & (HH - 1);
    float s = 0.0f;
    for (int p = 0; p < PAIR; ++p)
        s += (float)g[((size_t)b * PAIR + p) * HH + h];
    o[i] = (half_t)s;
}

__global__ __launch_bounds__(256)
void l2norm_kernel(const float* __restrict__ rec, float* __restrict__ out)
{
    int b = blockIdx.x;
    __shared__ float sh[256];
    float s = 0.0f;
    for (int h = threadIdx.x; h < HH; h += 256) { float v = rec[b * HH + h]; s += v * v; }
    sh[threadIdx.x] = s;
    __syncthreads();
    for (int st = 128; st; st >>= 1) {
        if (threadIdx.x < st) sh[threadIdx.x] += sh[threadIdx.x + st];
        __syncthreads();
    }
    float inv = rsqrtf(sh[0]);
    for (int h = threadIdx.x; h < HH; h += 256)
        out[b * HH + h] = rec[b * HH + h] * inv;
}

// =====================================================================
// Host orchestration
// =====================================================================
static inline void gemm(hipStream_t st, const half_t* A, const half_t* Bm,
                        float* Cf, half_t* Ch, const float* bias,
                        int M, int Nreal, int Npad, int Kd, int ldc, int act)
{
    int kch = (Kd > 320) ? 256 : Kd;
    size_t shb = (size_t)64 * (kch + 8) * sizeof(half_t);
    dim3 g(Npad / 64, M / 128), b(128);
    gemm_nt_wmma<<<g, b, shb, st>>>(A, Bm, Cf, Ch, bias, M, Nreal, Kd, ldc, act);
}

extern "C" void kernel_launch(void* const* d_in, const int* in_sizes, int n_in,
                              void* d_out, int out_size, void* d_ws, size_t ws_size,
                              hipStream_t stream)
{
    // ---- inputs (setup_inputs dict order) ----
    const float* img    = (const float*)d_in[0];
    const int*   toks   = (const int*)  d_in[1];
    const float* qword  = (const float*)d_in[2];
    const float* WihF   = (const float*)d_in[3];
    const float* WhhF   = (const float*)d_in[4];
    const float* bihF   = (const float*)d_in[5];
    const float* bhhF   = (const float*)d_in[6];
    const float* WihB   = (const float*)d_in[7];
    const float* WhhB   = (const float*)d_in[8];
    const float* bihB   = (const float*)d_in[9];
    const float* bhhB   = (const float*)d_in[10];
    const float* projW  = (const float*)d_in[11];
    const float* projb  = (const float*)d_in[12];
    const float* vattvW = (const float*)d_in[13];
    const float* vattvb = (const float*)d_in[14];
    const float* vattqW = (const float*)d_in[15];
    const float* vattqb = (const float*)d_in[16];
    const float* vattoW = (const float*)d_in[17];
    const float* vattob = (const float*)d_in[18];
    const float* qnetW  = (const float*)d_in[19];
    const float* qnetb  = (const float*)d_in[20];
    const float* vnetW  = (const float*)d_in[21];
    const float* vnetb  = (const float*)d_in[22];
    const float* clsW1  = (const float*)d_in[23];
    const float* clsb1  = (const float*)d_in[24];
    const float* clsW2  = (const float*)d_in[25];
    const float* clsb2  = (const float*)d_in[26];
    const float* g0W    = (const float*)d_in[27];
    const float* g0b    = (const float*)d_in[28];
    const float* g1W    = (const float*)d_in[29];
    const float* g1b    = (const float*)d_in[30];
    const float* g2W    = (const float*)d_in[31];
    const float* g2b    = (const float*)d_in[32];
    const float* g3W    = (const float*)d_in[33];
    const float* g3b    = (const float*)d_in[34];
    const float* f0W    = (const float*)d_in[35];
    const float* f0b    = (const float*)d_in[36];
    const float* f1W    = (const float*)d_in[37];
    const float* f1b    = (const float*)d_in[38];

    // ---- workspace arena ----
    char* ws = (char*)d_ws;
    size_t off = 0;
    auto alloc = [&](size_t bytes) -> char* {
        char* p = ws + off;
        off = (off + bytes + 255) & ~(size_t)255;
        return p;
    };

    half_t* hWihF  = (half_t*)alloc((size_t)G4 * EP * 2);
    half_t* hWihB  = (half_t*)alloc((size_t)G4 * EP * 2);
    half_t* hWhhF  = (half_t*)alloc((size_t)G4 * HH * 2);
    half_t* hWhhB  = (half_t*)alloc((size_t)G4 * HH * 2);
    half_t* hProj  = (half_t*)alloc((size_t)HH * 1024 * 2);
    half_t* hVW    = (half_t*)alloc((size_t)HH * HH * 2);
    half_t* hQW    = (half_t*)alloc((size_t)HH * HH * 2);
    half_t* hQnet  = (half_t*)alloc((size_t)HH * HH * 2);
    half_t* hVnet  = (half_t*)alloc((size_t)HH * HH * 2);
    half_t* hCls1  = (half_t*)alloc((size_t)1024 * HH * 2);
    half_t* hCls2  = (half_t*)alloc((size_t)VP * 1024 * 2);
    half_t* hG0    = (half_t*)alloc((size_t)HH * 1024 * 2);
    half_t* hG1    = (half_t*)alloc((size_t)HH * HH * 2);
    half_t* hG2    = (half_t*)alloc((size_t)HH * HH * 2);
    half_t* hG3    = (half_t*)alloc((size_t)HH * HH * 2);
    half_t* hF0    = (half_t*)alloc((size_t)HH * HH * 2);
    half_t* hF1    = (half_t*)alloc((size_t)HH * HH * 2);

    half_t* wemb   = (half_t*)alloc((size_t)NL * EP * 2);
    half_t* imgH   = (half_t*)alloc((size_t)NK * HH * 2);

    // big region shared by gates_x (NL*G4, live during LSTM) and vproj (NK*HH, live after)
    size_t bigBytes = (size_t)NK * HH * 2;   // 77MB >= NL*G4*2 (63MB)
    char*  bigreg   = alloc(bigBytes);
    half_t* gatesX  = (half_t*)bigreg;
    half_t* vproj   = (half_t*)bigreg;

    float*  gatesH = (float*) alloc((size_t)NN * G4 * 4);
    half_t* hF_    = (half_t*)alloc((size_t)NN * HH * 2);
    half_t* hB_    = (half_t*)alloc((size_t)NN * HH * 2);
    float*  cBuf   = (float*) alloc((size_t)NN * HH * 4);
    half_t* hcat   = (half_t*)alloc((size_t)NN * 1024 * 2);
    half_t* qemb   = (half_t*)alloc((size_t)NN * HH * 2);
    half_t* qproj  = (half_t*)alloc((size_t)NN * HH * 2);
    float*  attlog = (float*) alloc((size_t)NK * 4);
    half_t* vemb   = (half_t*)alloc((size_t)NN * HH * 2);
    half_t* qrep   = (half_t*)alloc((size_t)NN * HH * 2);
    half_t* vrep   = (half_t*)alloc((size_t)NN * HH * 2);
    half_t* joint  = (half_t*)alloc((size_t)NN * HH * 2);
    half_t* clshid = (half_t*)alloc((size_t)NN * 1024 * 2);
    half_t* catB   = (half_t*)alloc((size_t)NP * 1024 * 2);
    half_t* ga     = (half_t*)alloc((size_t)NP * HH * 2);
    half_t* gb     = (half_t*)alloc((size_t)NP * HH * 2);
    half_t* gsum   = (half_t*)alloc((size_t)BB * HH * 2);
    half_t* frep   = (half_t*)alloc((size_t)BB * HH * 2);
    float*  recf   = (float*) alloc((size_t)BB * HH * 4);
    (void)ws_size; (void)n_in; (void)in_sizes; (void)out_size;

    auto cvt = [&](const float* src, half_t* dst, int rows, int cols, int cols_pad, int rows_pad) {
        long long total = (long long)rows_pad * cols_pad;
        int blocks = (int)((total + 255) / 256);
        cvt_pad_kernel<<<blocks, 256, 0, stream>>>(src, dst, rows, cols, cols_pad, total);
    };

    // ---- weight / input conversions ----
    cvt(WihF,  hWihF, G4, EE, EP, G4);
    cvt(WihB,  hWihB, G4, EE, EP, G4);
    cvt(WhhF,  hWhhF, G4, HH, HH, G4);
    cvt(WhhB,  hWhhB, G4, HH, HH, G4);
    cvt(projW, hProj, HH, 1024, 1024, HH);
    cvt(vattvW,hVW,   HH, HH, HH, HH);
    cvt(vattqW,hQW,   HH, HH, HH, HH);
    cvt(qnetW, hQnet, HH, HH, HH, HH);
    cvt(vnetW, hVnet, HH, HH, HH, HH);
    cvt(clsW1, hCls1, 1024, HH, HH, 1024);
    cvt(clsW2, hCls2, VV, 1024, 1024, VP);
    cvt(g0W,   hG0,   HH, 1024, 1024, HH);
    cvt(g1W,   hG1,   HH, HH, HH, HH);
    cvt(g2W,   hG2,   HH, HH, HH, HH);
    cvt(g3W,   hG3,   HH, HH, HH, HH);
    cvt(f0W,   hF0,   HH, HH, HH, HH);
    cvt(f1W,   hF1,   HH, HH, HH, HH);
    cvt(img,   imgH,  NK, HH, HH, NK);

    {   // embedding gather + pad  [NL, EP]
        long long total = (long long)NL * EP;
        embed_kernel<<<(int)((total + 255) / 256), 256, 0, stream>>>(toks, qword, wemb, total);
    }

    // ---- bidirectional LSTM ----
    const int gateBlocks = (NN * HH + 255) / 256;
    for (int dir = 0; dir < 2; ++dir) {
        const half_t* Wih  = dir == 0 ? hWihF : hWihB;
        const half_t* Whh  = dir == 0 ? hWhhF : hWhhB;
        const float*  bih  = dir == 0 ? bihF : bihB;
        const float*  bhh  = dir == 0 ? bhhF : bhhB;
        half_t*       hCur = dir == 0 ? hF_ : hB_;

        // gates_x = w_emb @ Wih^T    [NL, 2048]  (f16 out)
        gemm(stream, wemb, Wih, nullptr, gatesX, nullptr, NL, G4, G4, EP, G4, 0);

        long long nh = (long long)NN * HH;
        zero16_kernel<<<(int)((nh + 255) / 256), 256, 0, stream>>>(hCur, nh);
        zero32_kernel<<<(int)((nh + 255) / 256), 256, 0, stream>>>(cBuf, nh);

        for (int t = 0; t < LL; ++t) {
            int tt = dir == 0 ? t : (LL - 1 - t);
            // gates_h = h @ Whh^T   [NN, 2048] (f32 out)
            gemm(stream, hCur, Whh, gatesH, nullptr, nullptr, NN, G4, G4, HH, G4, 0);
            lstm_gate_kernel<<<gateBlocks, 256, 0, stream>>>(gatesX, gatesH, bih, bhh, cBuf, hCur, tt);
        }
    }

    {   // hcat = [h_f, h_b]   [NN, 1024]
        long long total = (long long)NN * 1024;
        hcat_kernel<<<(int)((total + 255) / 256), 256, 0, stream>>>(hF_, hB_, hcat);
    }

    // q_emb = hcat @ proj_W^T + proj_b         [NN, 512]
    gemm(stream, hcat, hProj, nullptr, qemb, projb, NN, HH, HH, 1024, HH, 0);

    // v_proj = relu(img @ vatt_vW^T + vb)      [NK, 512]   (overwrites gates_x region)
    gemm(stream, imgH, hVW, nullptr, vproj, vattvb, NK, HH, HH, HH, HH, 1);

    // q_proj = relu(q_emb @ vatt_qW^T + qb)    [NN, 512]
    gemm(stream, qemb, hQW, nullptr, qproj, vattqb, NN, HH, HH, HH, HH, 1);

    // attention logits, softmax, weighted image sum
    attlogit_kernel<<<NK / 8, 256, 0, stream>>>(vproj, qproj, vattoW, vattob, attlog);
    softmax_vemb_kernel<<<NN, 256, 0, stream>>>(attlog, img, vemb);

    // joint = relu(q_emb@qnet^T+b) * relu(v_emb@vnet^T+b)
    gemm(stream, qemb, hQnet, nullptr, qrep, qnetb, NN, HH, HH, HH, HH, 1);
    gemm(stream, vemb, hVnet, nullptr, vrep, vnetb, NN, HH, HH, HH, HH, 1);
    {
        long long total = (long long)NN * HH;
        mul_kernel<<<(int)((total + 255) / 256), 256, 0, stream>>>(qrep, vrep, joint, total);
    }

    // classifier: logits -> d_out[0 : NN*VV]
    gemm(stream, joint, hCls1, nullptr, clshid, clsb1, NN, 1024, 1024, HH, 1024, 1);
    gemm(stream, clshid, hCls2, (float*)d_out, nullptr, clsb2, NN, VV, VP, 1024, VV, 0);

    // relation network
    {
        long long total = (long long)NP * 1024;
        cat_kernel<<<(int)((total + 255) / 256), 256, 0, stream>>>(joint, catB);
    }
    gemm(stream, catB, hG0, nullptr, ga, g0b, NP, HH, HH, 1024, HH, 1);
    gemm(stream, ga,   hG1, nullptr, gb, g1b, NP, HH, HH, HH, HH, 1);
    gemm(stream, gb,   hG2, nullptr, ga, g2b, NP, HH, HH, HH, HH, 1);
    gemm(stream, ga,   hG3, nullptr, gb, g3b, NP, HH, HH, HH, HH, 1);
    gsum_kernel<<<(BB * HH + 255) / 256, 256, 0, stream>>>(gb, gsum);

    gemm(stream, gsum, hF0, nullptr, frep, f0b, BB, HH, HH, HH, HH, 1);
    gemm(stream, frep, hF1, recf, nullptr, f1b, BB, HH, HH, HH, HH, 0);

    float* recOut = (float*)d_out + (size_t)NN * VV;
    l2norm_kernel<<<BB, 256, 0, stream>>>(recf, recOut);
}
